// SimpleGATLayer_27848567947366
// MI455X (gfx1250) — matrix operand order repaired and checked by reading, not compiled, not run on hardware
//
#include <hip/hip_runtime.h>
#include <hip/hip_bf16.h>
#include <stdint.h>

#define N_NODES 4096
#define IN_FEAT 256
#define OUT_FEAT 64
#define HEADS 4
#define NEG_SLOPE 0.2f
#define JT 32                 // j-tile (WMMA K for bf16)
#define NEG_BIG -1.0e30f

typedef __attribute__((ext_vector_type(16))) __bf16 v16bf;
typedef __attribute__((ext_vector_type(8)))  float  v8f;
typedef __attribute__((ext_vector_type(4)))  int    v4i;

#define AS1 __attribute__((address_space(1)))
#define AS3 __attribute__((address_space(3)))

#if __has_builtin(__builtin_amdgcn_global_load_async_to_lds_b128) && \
    __has_builtin(__builtin_amdgcn_s_wait_asynccnt)
#define USE_ASYNC 1
#else
#define USE_ASYNC 0
#endif

union AFrag { v16bf v; unsigned int u[8]; };

// cheap f32->bf16 (round-half-up): 1 add + shift/perm
static __device__ __forceinline__ unsigned short f2bf(float f) {
    return (unsigned short)((__float_as_uint(f) + 0x8000u) >> 16);
}
// pack two f32 -> packed bf16x2 with a single v_perm_b32
static __device__ __forceinline__ unsigned int pack_bf2(float lo, float hi) {
    const unsigned ul = __float_as_uint(lo) + 0x8000u;
    const unsigned uh = __float_as_uint(hi) + 0x8000u;
    // out bytes: [lo.b2, lo.b3, hi.b2, hi.b3]; S0=uh (bytes 7:4), S1=ul (3:0)
    return __builtin_amdgcn_perm(uh, ul, 0x07060302u);
}
static __device__ __forceinline__ float bf_lo(unsigned int u) {
    return __uint_as_float(u << 16);
}
static __device__ __forceinline__ float bf_hi(unsigned int u) {
    return __uint_as_float(u & 0xFFFF0000u);
}

// ---------------------------------------------------------------------------
// Kernel 0: stream-convert x and W to bf16 once (keeps conversions out of the
// WMMA loops). 4 elements / thread.
// ---------------------------------------------------------------------------
#define NX_ELEMS (N_NODES * IN_FEAT)                    // 1048576
#define NW_ELEMS (HEADS * IN_FEAT * OUT_FEAT)           // 65536
#define CVT_THREADS ((NX_ELEMS + NW_ELEMS) / 4)         // 278528

__global__ __launch_bounds__(256) void gat_cvt_kernel(
    const float* __restrict__ x, const float* __restrict__ W,
    unsigned short* __restrict__ xb, unsigned short* __restrict__ wb) {
    const int id = blockIdx.x * blockDim.x + threadIdx.x;
    if (id >= CVT_THREADS) return;
    const float* srcp;
    unsigned int* dstp;
    if (id < NX_ELEMS / 4) {
        srcp = x + (size_t)id * 4;
        dstp = (unsigned int*)xb + (size_t)id * 2;
    } else {
        const int j = id - NX_ELEMS / 4;
        srcp = W + (size_t)j * 4;
        dstp = (unsigned int*)wb + (size_t)j * 2;
    }
    const float4 v = *(const float4*)srcp;
    dstp[0] = pack_bf2(v.x, v.y);
    dstp[1] = pack_bf2(v.z, v.w);
}

// ---------------------------------------------------------------------------
// Kernel 1: xt[h,n,f] = sum_i x[n,i] * W[h,i,f]   (bf16 WMMA, f32 accum)
// One wave = one (head, 16-row tile). Block = 4 waves = 4 heads, same rows.
// A-frag (16x32 bf16): lane = M row (mod 16), VGPR v holds K pair
//   kv = (v<4 ? 2v : 16+2(v-4)) + 8*(lane>=16)
// B-frag (32x16 bf16): lane = N col (mod 16), element s holds K in same order as A
// C/D (16x16 f32): VGPR r: lanes0-15 -> M=r, lanes16-31 -> M=r+8; N = lane&15
// ---------------------------------------------------------------------------
__global__ __launch_bounds__(128) void gat_xt_kernel(
    const unsigned short* __restrict__ xb, const unsigned short* __restrict__ wb,
    unsigned short* __restrict__ xtb) {
    const int lane = threadIdx.x & 31;
    const int h    = threadIdx.x >> 5;
    const int R    = blockIdx.x * 16;
    const int row  = lane & 15;
    const int half = lane >> 4;

    int kv[8];
#pragma unroll
    for (int v = 0; v < 8; ++v)
        kv[v] = ((v < 4) ? (2 * v) : (16 + 2 * (v - 4))) + 8 * half;

    const v8f z8 = {0.f,0.f,0.f,0.f,0.f,0.f,0.f,0.f};
    v8f acc[4] = {z8, z8, z8, z8};

    const unsigned short* xrow = xb + (size_t)(R + row) * IN_FEAT;
    const unsigned short* wh   = wb + (size_t)h * IN_FEAT * OUT_FEAT;

    for (int K = 0; K < IN_FEAT; K += JT) {
        AFrag af;
#pragma unroll
        for (int v = 0; v < 8; ++v)
            af.u[v] = *(const unsigned int*)(xrow + K + kv[v]);   // bf16 pair

        const unsigned short* wrow = wh + (size_t)K * OUT_FEAT;
#pragma unroll
        for (int nb = 0; nb < 4; ++nb) {
            v16bf bfrag;
#pragma unroll
            for (int s = 0; s < 16; ++s)
                bfrag[s] = __builtin_bit_cast(__bf16, wrow[(size_t)(((s < 8) ? s : (8 + s)) + 8 * half) * OUT_FEAT + nb * 16 + row]);
            acc[nb] = __builtin_amdgcn_wmma_f32_16x16x32_bf16(
                false, af.v, false, bfrag, (short)0, acc[nb], false, false);
        }
    }

#pragma unroll
    for (int r = 0; r < 8; ++r) {
        const int m = r + 8 * half;
#pragma unroll
        for (int nb = 0; nb < 4; ++nb) {
            const size_t idx =
                ((size_t)h * N_NODES + (R + m)) * OUT_FEAT + nb * 16 + row;
            xtb[idx] = f2bf(acc[nb][r]);
        }
    }
}

// ---------------------------------------------------------------------------
// Kernel 2: src[h,n] = xt[h,n,:] . a[h,:64];  dst[h,n] = xt[h,n,:] . a[h,64:]
// reads bf16 xt (shift-expand), one thread per (h,n)
// ---------------------------------------------------------------------------
__global__ __launch_bounds__(256) void gat_srcdst_kernel(
    const unsigned short* __restrict__ xtb, const float* __restrict__ a,
    float* __restrict__ src, float* __restrict__ dst) {
    const int id = blockIdx.x * blockDim.x + threadIdx.x;
    if (id >= HEADS * N_NODES) return;
    const int h = id / N_NODES;
    const unsigned int* xr = (const unsigned int*)(xtb + (size_t)id * OUT_FEAT);
    const float* ah = a + h * 2 * OUT_FEAT;
    float s = 0.f, d = 0.f;
#pragma unroll 8
    for (int f2 = 0; f2 < OUT_FEAT / 2; ++f2) {
        const unsigned int u = xr[f2];
        const float v0 = bf_lo(u), v1 = bf_hi(u);
        s += v0 * ah[2 * f2] + v1 * ah[2 * f2 + 1];
        d += v0 * ah[OUT_FEAT + 2 * f2] + v1 * ah[OUT_FEAT + 2 * f2 + 1];
    }
    src[id] = s;
    dst[id] = d;
}

// ---------------------------------------------------------------------------
// Kernel 3: fused masked-softmax attention + (attn @ xt), flash-style.
// Block = 4 waves (one head each) over the same 16-row tile; adj tile shared
// through LDS via GLOBAL_LOAD_ASYNC_TO_LDS_B128 (ASYNCcnt), double-buffered,
// one s_wait_asynccnt + one barrier per j-tile. Probabilities packed to a
// bf16 A-fragment in registers; xt rows load directly in B-fragment layout.
// ---------------------------------------------------------------------------
__global__ __launch_bounds__(128) void gat_attn_kernel(
    const int* __restrict__ adj, const unsigned short* __restrict__ xtb,
    const float* __restrict__ src, const float* __restrict__ dst,
    float* __restrict__ out) {
    __shared__ int s_adj[2][16][JT];

    const int lane = threadIdx.x & 31;
    const int h    = threadIdx.x >> 5;
    const int R    = blockIdx.x * 16;
    const int row  = lane & 15;
    const int half = lane >> 4;

    int kv[8];
#pragma unroll
    for (int v = 0; v < 8; ++v)
        kv[v] = ((v < 4) ? (2 * v) : (16 + 2 * (v - 4))) + 8 * half;

    const float s_i = src[h * N_NODES + R + row];
    const float* dh = dst + h * N_NODES;
    const unsigned short* xth = xtb + (size_t)h * N_NODES * OUT_FEAT;

    float m    = NEG_BIG;
    float lsum = 0.f;
    const v8f z8 = {0.f,0.f,0.f,0.f,0.f,0.f,0.f,0.f};
    v8f acc[4] = {z8, z8, z8, z8};

    // cooperative adj staging: 128 threads x 16B = 16x32 ints per tile
    const int srow = threadIdx.x >> 3;
    const int scol = (threadIdx.x & 7) * 4;
    const int* gadj = adj + (size_t)(R + srow) * N_NODES + scol;
    const int NT = N_NODES / JT;                       // 128 tiles

#if USE_ASYNC
    // issue tile 0
    __builtin_amdgcn_global_load_async_to_lds_b128(
        (AS1 v4i*)(uintptr_t)gadj,
        (AS3 v4i*)(void*)&s_adj[0][srow][scol], 0, 0);
#else
    int4 pre = *(const int4*)gadj;
#endif

    for (int t = 0; t < NT; ++t) {
        const int buf = t & 1;
#if USE_ASYNC
        __builtin_amdgcn_s_wait_asynccnt(0);           // my tile-t piece landed
        __syncthreads();                               // everyone's landed; all
                                                       // reads of buf^1 done
        if (t + 1 < NT)
            __builtin_amdgcn_global_load_async_to_lds_b128(
                (AS1 v4i*)(uintptr_t)(gadj + (size_t)(t + 1) * JT),
                (AS3 v4i*)(void*)&s_adj[buf ^ 1][srow][scol], 0, 0);
#else
        *(int4*)&s_adj[buf][srow][scol] = pre;
        __syncthreads();
        if (t + 1 < NT)
            pre = *(const int4*)(gadj + (size_t)(t + 1) * JT);
#endif
        const int J = t * JT;

        // masked leaky-relu logits for this lane's 16 (row, J+kv) entries
        float ev[16];
        float mt = NEG_BIG;
#pragma unroll
        for (int v = 0; v < 8; ++v) {
            const float2 dd = *(const float2*)(dh + J + kv[v]);
            const int2   aa = *(const int2*)&s_adj[buf][row][kv[v]];
            float e0 = s_i + dd.x; e0 = fmaxf(e0, NEG_SLOPE * e0);
            float e1 = s_i + dd.y; e1 = fmaxf(e1, NEG_SLOPE * e1);
            e0 = (aa.x > 0) ? e0 : NEG_BIG;
            e1 = (aa.y > 0) ? e1 : NEG_BIG;
            ev[2 * v]     = e0;
            ev[2 * v + 1] = e1;
            mt = fmaxf(mt, fmaxf(e0, e1));
        }
        // combine the two K-halves of each row (lanes l and l^16)
        mt = fmaxf(mt, __shfl_xor(mt, 16, 32));
        const float mnew  = fmaxf(m, mt);
        const float scale = __expf(m - mnew); // ==1 if unchanged, ->0 if jumped
        m = mnew;
        lsum *= scale;

        AFrag pf;
#pragma unroll
        for (int v = 0; v < 8; ++v) {
            const float p0 = (ev[2*v]   > -1.0e29f) ? __expf(ev[2*v]   - mnew) : 0.f;
            const float p1 = (ev[2*v+1] > -1.0e29f) ? __expf(ev[2*v+1] - mnew) : 0.f;
            lsum += p0 + p1;
            pf.u[v] = pack_bf2(p0, p1);
        }

        // rescale accumulators: per-row scale broadcast (row state lives in
        // lanes r / r+8 of the low half-wave)
#pragma unroll
        for (int r = 0; r < 8; ++r) {
            const float sr = __shfl(scale, r + 8 * half, 32);
            acc[0][r] *= sr; acc[1][r] *= sr; acc[2][r] *= sr; acc[3][r] *= sr;
        }

        // B fragments: lane = N col (mod 16) of xt, element s holds K in same order as A
        const unsigned short* xrow = xth + (size_t)J * OUT_FEAT;
#pragma unroll
        for (int nb = 0; nb < 4; ++nb) {
            v16bf bfrag;
#pragma unroll
            for (int s = 0; s < 16; ++s)
                bfrag[s] = __builtin_bit_cast(__bf16, xrow[(size_t)(((s < 8) ? s : (8 + s)) + 8 * half) * OUT_FEAT + nb * 16 + row]);
            acc[nb] = __builtin_amdgcn_wmma_f32_16x16x32_bf16(
                false, pf.v, false, bfrag, (short)0, acc[nb], false, false);
        }
    }

    // finalize: 1/rowsum, normalize, write out[n, h*64 + f]
    lsum += __shfl_xor(lsum, 16, 32);
    const float linv = (lsum > 0.f) ? (1.f / lsum) : 0.f;
#pragma unroll
    for (int r = 0; r < 8; ++r) {
        const float li  = __shfl(linv, r + 8 * half, 32);
        const int mrow  = R + r + 8 * half;
        float* orow = out + (size_t)mrow * (HEADS * OUT_FEAT) + h * OUT_FEAT;
#pragma unroll
        for (int nb = 0; nb < 4; ++nb)
            orow[nb * 16 + row] = acc[nb][r] * li;
    }
}

// ---------------------------------------------------------------------------
extern "C" void kernel_launch(void* const* d_in, const int* in_sizes, int n_in,
                              void* d_out, int out_size, void* d_ws, size_t ws_size,
                              hipStream_t stream) {
    (void)in_sizes; (void)n_in; (void)out_size; (void)ws_size;
    const float* x   = (const float*)d_in[0];
    const float* W   = (const float*)d_in[1];
    const float* a   = (const float*)d_in[2];
    const int*   adj = (const int*)d_in[3];
    float* out = (float*)d_out;

    char* ws = (char*)d_ws;
    const size_t xt_elems = (size_t)HEADS * N_NODES * OUT_FEAT;    // 1M
    unsigned short* xtb = (unsigned short*)ws;                      // 2 MB
    unsigned short* xbb = (unsigned short*)(ws + xt_elems * 2);     // 2 MB
    unsigned short* wbb = (unsigned short*)(ws + xt_elems * 2 + (size_t)NX_ELEMS * 2); // 128KB
    float* src = (float*)(ws + xt_elems * 2 + (size_t)(NX_ELEMS + NW_ELEMS) * 2);
    float* dst = src + (size_t)HEADS * N_NODES;

    gat_cvt_kernel<<<(CVT_THREADS + 255) / 256, 256, 0, stream>>>(x, W, xbb, wbb);
    gat_xt_kernel<<<N_NODES / 16, 128, 0, stream>>>(xbb, wbb, xtb);
    gat_srcdst_kernel<<<(HEADS * N_NODES) / 256, 256, 0, stream>>>(xtb, a, src, dst);
    gat_attn_kernel<<<N_NODES / 16, 128, 0, stream>>>(adj, xtb, src, dst, out);
}
